// ArcticMoeBlock_6073083756875
// MI455X (gfx1250) — matrix operand order repaired and checked by reading, not compiled, not run on hardware
//
#include <hip/hip_runtime.h>
#include <hip/hip_bf16.h>

#define S_TOK 2048
#define H_DIM 1024
#define F_DIM 4096
#define E_NUM 8

typedef __attribute__((ext_vector_type(16))) __bf16 v16bf;
typedef __attribute__((ext_vector_type(8)))  float  v8f;
typedef __attribute__((ext_vector_type(4)))  float  v4f;
typedef __attribute__((ext_vector_type(4)))  unsigned int u32x4;
typedef __attribute__((ext_vector_type(2)))  unsigned int u32x2;

union Frag16 {
    v16bf bf;
    u32x4 u[2];
};

// fp32 -> bf16 with round-to-nearest-even
static __device__ __forceinline__ unsigned short f2bf(float f) {
    unsigned int u = __float_as_uint(f);
    u += 0x7FFFu + ((u >> 16) & 1u);
    return (unsigned short)(u >> 16);
}

static __device__ __forceinline__ unsigned int pack2bf(float a, float b) {
    return (unsigned int)f2bf(a) | ((unsigned int)f2bf(b) << 16);
}

static __device__ __forceinline__ float fast_sigmoid(float z) {
    // 1 / (1 + e^-z) via v_rcp_f32 (avoid IEEE div expansion)
    return __builtin_amdgcn_rcpf(1.f + __expf(-z));
}

// ---------------------------------------------------------------------------
// Router: logits = x @ Wg  -> top-2 -> softmax over the 2 -> combine[S,E]
// One wave (32 lanes) per token.
// ---------------------------------------------------------------------------
__global__ __launch_bounds__(256)
void moe_router_kernel(const float* __restrict__ x, const float* __restrict__ Wg,
                       float* __restrict__ logits_out, float* __restrict__ combine) {
    int token = blockIdx.x * 8 + (threadIdx.x >> 5);
    int lane  = threadIdx.x & 31;

    float acc[E_NUM];
#pragma unroll
    for (int e = 0; e < E_NUM; ++e) acc[e] = 0.f;

    for (int h = lane; h < H_DIM; h += 32) {
        float xv = x[(size_t)token * H_DIM + h];
#pragma unroll
        for (int e = 0; e < E_NUM; ++e)
            acc[e] += xv * Wg[h * E_NUM + e];
    }
    // wave32 reduction
#pragma unroll
    for (int off = 16; off > 0; off >>= 1) {
#pragma unroll
        for (int e = 0; e < E_NUM; ++e)
            acc[e] += __shfl_xor(acc[e], off, 32);
    }
    if (lane == 0) {
        int i1 = 0;
#pragma unroll
        for (int e = 1; e < E_NUM; ++e) if (acc[e] > acc[i1]) i1 = e;
        int i2 = (i1 == 0) ? 1 : 0;
#pragma unroll
        for (int e = 0; e < E_NUM; ++e)
            if (e != i1 && acc[e] > acc[i2]) i2 = e;
        // softmax over the two selected logits
        float p1 = fast_sigmoid(acc[i1] - acc[i2]);
        float p2 = 1.f - p1;
#pragma unroll
        for (int e = 0; e < E_NUM; ++e) {
            logits_out[token * E_NUM + e] = acc[e];
            combine[token * E_NUM + e] = 0.f;
        }
        combine[token * E_NUM + i1] = p1;
        combine[token * E_NUM + i2] = p2;
    }
}

// ---------------------------------------------------------------------------
// Pass A (per expert): Hbuf[s,f] = silu(x@W1e)[s,f] * (x@W3e)[s,f]  (bf16 out)
// Tile 128(M=S) x 128(N=F), K = H in steps of 32. 8 waves; each wave 64x32.
// ---------------------------------------------------------------------------
#define KPAD 40   // bf16 elements per LDS row (32 data + 8 pad), 80B = 16B aligned

__global__ __launch_bounds__(256)
void moe_h_kernel(const float* __restrict__ x,
                  const float* __restrict__ W1e,
                  const float* __restrict__ W3e,
                  unsigned short* __restrict__ Hbuf) {
    __shared__ __align__(16) unsigned short As[128 * KPAD];
    __shared__ __align__(16) unsigned short B1s[128 * KPAD];
    __shared__ __align__(16) unsigned short B3s[128 * KPAD];

    const int t     = threadIdx.x;
    const int wave  = t >> 5;
    const int lane  = t & 31;
    const int wm    = wave >> 2;       // 0..1  (64-row slabs)
    const int wn    = wave & 3;        // 0..3  (32-col slabs)
    const int m16   = lane & 15;
    const int half  = lane >> 4;
    const int tileM = blockIdx.y * 128;    // over S
    const int tileN = blockIdx.x * 128;    // over F

    v8f acc1[4][2], acc3[4][2];
#pragma unroll
    for (int fm = 0; fm < 4; ++fm)
#pragma unroll
        for (int fn = 0; fn < 2; ++fn) { acc1[fm][fn] = {}; acc3[fm][fn] = {}; }

    for (int kk = 0; kk < H_DIM; kk += 32) {
        // x tile: 128 rows x 32 cols fp32, b128 loads, pack 4 bf16 -> b64 LDS store
#pragma unroll
        for (int i = 0; i < 4; ++i) {
            int chunk = t + i * 256;            // 1024 float4 chunks
            int r = chunk >> 3, c4 = (chunk & 7) * 4;
            v4f v = *(const v4f*)&x[(size_t)(tileM + r) * H_DIM + kk + c4];
            u32x2 pk;
            pk.x = pack2bf(v.x, v.y);
            pk.y = pack2bf(v.z, v.w);
            *(u32x2*)&As[r * KPAD + c4] = pk;
        }
        // W tiles stored N-major: B[f_local][k]; b128 global loads + prefetch
#pragma unroll
        for (int i = 0; i < 4; ++i) {
            int chunk = t + i * 256;            // 1024 float4 chunks
            int k = chunk >> 5, f4 = (chunk & 31) * 4;
            size_t g = (size_t)(kk + k) * F_DIM + tileN + f4;
            v4f w1 = *(const v4f*)&W1e[g];
            v4f w3 = *(const v4f*)&W3e[g];
#pragma unroll
            for (int j = 0; j < 4; ++j) {
                B1s[(f4 + j) * KPAD + k] = f2bf(w1[j]);
                B3s[(f4 + j) * KPAD + k] = f2bf(w3[j]);
            }
            if (kk + 32 < H_DIM) {
                __builtin_prefetch(&W1e[g + (size_t)32 * F_DIM], 0, 0);
                __builtin_prefetch(&W3e[g + (size_t)32 * F_DIM], 0, 0);
            }
        }
        __syncthreads();

        Frag16 a[4];
#pragma unroll
        for (int fm = 0; fm < 4; ++fm) {
            int row = wm * 64 + fm * 16 + m16;
            const unsigned short* p = &As[row * KPAD + half * 8];
            a[fm].u[0] = *(const u32x4*)p;          // K = half*8 .. +7
            a[fm].u[1] = *(const u32x4*)(p + 16);   // K = 16+half*8 .. +7
        }
        Frag16 b1[2], b3[2];
#pragma unroll
        for (int fn = 0; fn < 2; ++fn) {
            int col = wn * 32 + fn * 16 + m16;
            const unsigned short* p1 = &B1s[col * KPAD + half * 8];
            b1[fn].u[0] = *(const u32x4*)p1;        // K = half*8 .. +7
            b1[fn].u[1] = *(const u32x4*)(p1 + 16); // K = 16+half*8 .. +7
            const unsigned short* p3 = &B3s[col * KPAD + half * 8];
            b3[fn].u[0] = *(const u32x4*)p3;
            b3[fn].u[1] = *(const u32x4*)(p3 + 16);
        }
#pragma unroll
        for (int fm = 0; fm < 4; ++fm)
#pragma unroll
            for (int fn = 0; fn < 2; ++fn) {
                acc1[fm][fn] = __builtin_amdgcn_wmma_f32_16x16x32_bf16(
                    false, a[fm].bf, false, b1[fn].bf, (short)0, acc1[fm][fn], false, false);
                acc3[fm][fn] = __builtin_amdgcn_wmma_f32_16x16x32_bf16(
                    false, a[fm].bf, false, b3[fn].bf, (short)0, acc3[fm][fn], false, false);
            }
        __syncthreads();
    }

    // epilogue: silu(h1) * h3 -> bf16 Hbuf[s, f]
#pragma unroll
    for (int fm = 0; fm < 4; ++fm) {
        int baseRow = tileM + wm * 64 + fm * 16 + half * 8;
#pragma unroll
        for (int fn = 0; fn < 2; ++fn) {
            int col = tileN + wn * 32 + fn * 16 + m16;
#pragma unroll
            for (int v = 0; v < 8; ++v) {
                float h1 = acc1[fm][fn][v];
                float h3 = acc3[fm][fn][v];
                Hbuf[(size_t)(baseRow + v) * F_DIM + col] = f2bf(h1 * fast_sigmoid(h1) * h3);
            }
        }
    }
}

// ---------------------------------------------------------------------------
// Pass B (per expert): out[s,h] (+)= combine[s,e] * (Hbuf @ W2e)[s,h]
// Tile 128(M=S) x 128(N=H), K = F in steps of 32.
// A-tile (already bf16) is brought in with GLOBAL_LOAD_ASYNC_TO_LDS_B128.
// ---------------------------------------------------------------------------
__global__ __launch_bounds__(256)
void moe_out_kernel(const unsigned short* __restrict__ Hbuf,
                    const float* __restrict__ W2e,
                    const float* __restrict__ combine,
                    float* __restrict__ out,
                    int expert, int accumulate) {
    __shared__ __align__(16) unsigned short As[128 * KPAD];
    __shared__ __align__(16) unsigned short Bs[128 * KPAD];
    __shared__ float cwS[128];

    const int t     = threadIdx.x;
    const int wave  = t >> 5;
    const int lane  = t & 31;
    const int wm    = wave >> 2;
    const int wn    = wave & 3;
    const int m16   = lane & 15;
    const int half  = lane >> 4;
    const int tileM = blockIdx.y * 128;    // over S
    const int tileN = blockIdx.x * 128;    // over H

    if (t < 128) cwS[t] = combine[(tileM + t) * E_NUM + expert];

    v8f acc[4][2];
#pragma unroll
    for (int fm = 0; fm < 4; ++fm)
#pragma unroll
        for (int fn = 0; fn < 2; ++fn) acc[fm][fn] = {};

    for (int kk = 0; kk < F_DIM; kk += 32) {
        // A tile: 128 x 32 bf16 = 8 KB, async DMA to LDS (512 x 16B chunks)
#pragma unroll
        for (int i = 0; i < 2; ++i) {
            int chunk = t + i * 256;
            int r = chunk >> 2, q = (chunk & 3) * 8;
            unsigned ldsa = (unsigned)(uintptr_t)&As[r * KPAD + q];
            const unsigned short* g = Hbuf + (size_t)(tileM + r) * F_DIM + kk + q;
            asm volatile("global_load_async_to_lds_b128 %0, %1, off"
                         :: "v"(ldsa), "v"(g) : "memory");
        }
        // W2 tile, N-major: Bs[h_local][k], b128 loads, fp32 -> bf16
#pragma unroll
        for (int i = 0; i < 4; ++i) {
            int chunk = t + i * 256;            // 1024 float4 chunks
            int k = chunk >> 5, h4 = (chunk & 31) * 4;
            size_t g = (size_t)(kk + k) * H_DIM + tileN + h4;
            v4f w2 = *(const v4f*)&W2e[g];
#pragma unroll
            for (int j = 0; j < 4; ++j)
                Bs[(h4 + j) * KPAD + k] = f2bf(w2[j]);
            if (kk + 32 < F_DIM)
                __builtin_prefetch(&W2e[g + (size_t)32 * H_DIM], 0, 0);
        }
        asm volatile("s_wait_asynccnt 0" ::: "memory");
        __syncthreads();

        Frag16 a[4];
#pragma unroll
        for (int fm = 0; fm < 4; ++fm) {
            int row = wm * 64 + fm * 16 + m16;
            const unsigned short* p = &As[row * KPAD + half * 8];
            a[fm].u[0] = *(const u32x4*)p;
            a[fm].u[1] = *(const u32x4*)(p + 16);
        }
        Frag16 b[2];
#pragma unroll
        for (int fn = 0; fn < 2; ++fn) {
            int col = wn * 32 + fn * 16 + m16;
            const unsigned short* p = &Bs[col * KPAD + half * 8];
            b[fn].u[0] = *(const u32x4*)p;
            b[fn].u[1] = *(const u32x4*)(p + 16);
        }
#pragma unroll
        for (int fm = 0; fm < 4; ++fm)
#pragma unroll
            for (int fn = 0; fn < 2; ++fn)
                acc[fm][fn] = __builtin_amdgcn_wmma_f32_16x16x32_bf16(
                    false, a[fm].bf, false, b[fn].bf, (short)0, acc[fm][fn], false, false);
        __syncthreads();
    }

#pragma unroll
    for (int fm = 0; fm < 4; ++fm) {
        int localRow = wm * 64 + fm * 16 + half * 8;
#pragma unroll
        for (int fn = 0; fn < 2; ++fn) {
            int col = tileN + wn * 32 + fn * 16 + m16;
#pragma unroll
            for (int v = 0; v < 8; ++v) {
                float cw = cwS[localRow + v];
                size_t idx = (size_t)(tileM + localRow + v) * H_DIM + col;
                float val = cw * acc[fm][fn][v];
                out[idx] = accumulate ? (out[idx] + val) : val;
            }
        }
    }
}

// ---------------------------------------------------------------------------
extern "C" void kernel_launch(void* const* d_in, const int* in_sizes, int n_in,
                              void* d_out, int out_size, void* d_ws, size_t ws_size,
                              hipStream_t stream) {
    const float* x  = (const float*)d_in[0];   // [1,S,H]
    const float* Wg = (const float*)d_in[1];   // [H,E]
    const float* W1 = (const float*)d_in[2];   // [E,H,F]
    const float* W3 = (const float*)d_in[3];   // [E,H,F]
    const float* W2 = (const float*)d_in[4];   // [E,F,H]

    float* outFinal = (float*)d_out;                       // [S,H]
    float* logits   = outFinal + (size_t)S_TOK * H_DIM;    // [S,E]

    float* combine = (float*)d_ws;                          // S*E floats = 64 KB
    unsigned short* Hbuf = (unsigned short*)((char*)d_ws + 65536);  // S*F bf16 = 16 MB

    moe_router_kernel<<<S_TOK / 8, 256, 0, stream>>>(x, Wg, logits, combine);

    for (int e = 0; e < E_NUM; ++e) {
        const float* W1e = W1 + (size_t)e * H_DIM * F_DIM;
        const float* W3e = W3 + (size_t)e * H_DIM * F_DIM;
        const float* W2e = W2 + (size_t)e * F_DIM * H_DIM;

        dim3 gA(F_DIM / 128, S_TOK / 128);   // 32 x 16
        moe_h_kernel<<<gA, 256, 0, stream>>>(x, W1e, W3e, Hbuf);

        dim3 gB(H_DIM / 128, S_TOK / 128);   // 8 x 16
        moe_out_kernel<<<gB, 256, 0, stream>>>(Hbuf, W2e, combine, outFinal, e, e > 0);
    }
}
